// Block_22325240004804
// MI455X (gfx1250) — matrix operand order repaired and checked by reading, not compiled, not run on hardware
//
#include <hip/hip_runtime.h>

// ---------------------------------------------------------------------------
// CDNA5 (gfx1250, wave32) transformer cross-block.
// All GEMM-shaped math via v_wmma_f32_16x16x32_bf16 (bf16 in, f32 accum).
// GEMM: 32x64 output tile per wave -> 8 WMMA per K=32 step, 12 b128 loads.
// ---------------------------------------------------------------------------

typedef __attribute__((ext_vector_type(16))) __bf16 v16bf;
typedef __attribute__((ext_vector_type(8)))  float  v8f;
typedef __attribute__((ext_vector_type(4)))  unsigned int u32x4;

union BFrag { u32x4 u[2]; v16bf v; };

#define WMMA_BF16(Af, Bf, Cacc) \
    __builtin_amdgcn_wmma_f32_16x16x32_bf16(false, (Af), false, (Bf), (short)0, (Cacc), false, false)

// ------------------------- elementwise f32 -> bf16 -------------------------
__global__ __launch_bounds__(256) void cvt_bf16(const float* __restrict__ in,
                                                __bf16* __restrict__ out, int n) {
    int i = blockIdx.x * 256 + threadIdx.x;
    if (i < n) out[i] = (__bf16)in[i];
}

// ------------------------- row LayerNorm over 768 --------------------------
__global__ __launch_bounds__(256) void ln768(const float* __restrict__ x,
                                             const float* __restrict__ w,
                                             const float* __restrict__ bi,
                                             float* __restrict__ outf,
                                             __bf16* __restrict__ outb) {
    __shared__ float ssum[256], ssq[256];
    const int row = blockIdx.x, tid = threadIdx.x;
    const float* xr = x + (size_t)row * 768;
    float v0 = xr[tid], v1 = xr[tid + 256], v2 = xr[tid + 512];
    ssum[tid] = v0 + v1 + v2;
    ssq[tid]  = v0 * v0 + v1 * v1 + v2 * v2;
    __syncthreads();
    for (int st = 128; st > 0; st >>= 1) {
        if (tid < st) { ssum[tid] += ssum[tid + st]; ssq[tid] += ssq[tid + st]; }
        __syncthreads();
    }
    float mean = ssum[0] * (1.f / 768.f);
    float var  = ssq[0] * (1.f / 768.f) - mean * mean;
    float rstd = rsqrtf(var + 1e-5f);
#pragma unroll
    for (int i = 0; i < 3; i++) {
        int c = tid + i * 256;
        float xv = (i == 0 ? v0 : (i == 1 ? v1 : v2));
        float y = (xv - mean) * rstd * w[c] + bi[c];
        if (outf) outf[(size_t)row * 768 + c] = y;
        outb[(size_t)row * 768 + c] = (__bf16)y;
    }
}

// ---------------- per-head LayerNorm over head_dim=64 ----------------------
// qkv raw: (R=8192 rows) x 2304, col = t*768 + h*64 + d.
// t=0 -> q fp32 (B,H,N,D); t=1 -> k bf16 (B,H,N,D); t=2 -> v bf16 (B,H,D,N).
__global__ __launch_bounds__(256) void head_ln(const float* __restrict__ qkv,
                                               const float* __restrict__ hw,
                                               const float* __restrict__ hb,
                                               float* __restrict__ qf,
                                               __bf16* __restrict__ kb,
                                               __bf16* __restrict__ vb) {
    int g = blockIdx.x * 8 + (threadIdx.x >> 5);
    int lane = threadIdx.x & 31;
    int row = g / 36, rem = g % 36;
    int t = rem / 12, h = rem % 12;
    int b = row >> 10, ntok = row & 1023;
    const float* src = qkv + (size_t)row * 2304 + t * 768 + h * 64;
    float x0 = src[2 * lane], x1 = src[2 * lane + 1];
    float s = x0 + x1, q = x0 * x0 + x1 * x1;
#pragma unroll
    for (int m = 1; m < 32; m <<= 1) { s += __shfl_xor(s, m, 32); q += __shfl_xor(q, m, 32); }
    float mean = s * (1.f / 64.f), var = q * (1.f / 64.f) - mean * mean;
    float rstd = rsqrtf(var + 1e-5f);
    size_t bh = (size_t)(b * 12 + h);
#pragma unroll
    for (int i = 0; i < 2; i++) {
        int d = 2 * lane + i;
        float xv = (i == 0 ? x0 : x1);
        float y = (xv - mean) * rstd * hw[d] + hb[d];
        if (t == 0)      qf[(bh * 1024 + ntok) * 64 + d] = y;
        else if (t == 1) kb[(bh * 1024 + ntok) * 64 + d] = (__bf16)y;
        else             vb[(bh * 64 + d) * 1024 + ntok] = (__bf16)y;
    }
}

// ------------------- generic 32x64-per-wave WMMA GEMM ----------------------
// C[M,Nn] = A[M,K] (row-major bf16) @ W[Nn,K]^T (row-major bf16), f32 accum.
// MODE 0: plain f32 out          MODE 1: +bias +xn residual +q-reshape (proj)
// MODE 2: +bias, exact GELU, bf16 out (fc1)   MODE 3: out += gemm + bias (fc2)
template <int MODE>
__global__ __launch_bounds__(32) void gemm32x64(const __bf16* __restrict__ A,
                                                const __bf16* __restrict__ W,
                                                int M, int Nn, int K,
                                                float* __restrict__ outf,
                                                __bf16* __restrict__ outb,
                                                const float* __restrict__ bias,
                                                const float* __restrict__ resid,
                                                const float* __restrict__ qres) {
    const int lane = threadIdx.x, lo = lane & 15, hi = lane >> 4;
    const int m0 = blockIdx.x * 32, n0 = blockIdx.y * 64;
    v8f acc[2][4] = {};
    const __bf16* arow0 = A + (size_t)(m0 + lo) * K + 16 * hi;
    const __bf16* arow1 = arow0 + (size_t)16 * K;
    for (int k0 = 0; k0 < K; k0 += 32) {
        BFrag af[2];
        {
            const u32x4* ap0 = (const u32x4*)(arow0 + k0);
            const u32x4* ap1 = (const u32x4*)(arow1 + k0);
            af[0].u[0] = ap0[0]; af[0].u[1] = ap0[1];   // K = k0+16*hi+[0,16)
            af[1].u[0] = ap1[0]; af[1].u[1] = ap1[1];
        }
#pragma unroll
        for (int nt = 0; nt < 4; nt++) {
            const u32x4* wp = (const u32x4*)(W + (size_t)(n0 + 16 * nt + lo) * K + k0 + 16 * hi);
            BFrag wf;
            wf.u[0] = wp[0]; wf.u[1] = wp[1];           // K = k0+16*hi+[0,16)
            __builtin_prefetch((const void*)(wp + 4), 0, 3);  // next K-chunk of this W row
            acc[0][nt] = WMMA_BF16(af[0].v, wf.v, acc[0][nt]);
            acc[1][nt] = WMMA_BF16(af[1].v, wf.v, acc[1][nt]);
        }
    }
#pragma unroll
    for (int mi = 0; mi < 2; mi++) {
#pragma unroll
        for (int nt = 0; nt < 4; nt++) {
#pragma unroll
            for (int r = 0; r < 8; r++) {
                int row = m0 + 16 * mi + r + 8 * hi, col = n0 + 16 * nt + lo;
                float vacc = acc[mi][nt][r];
                if (MODE == 0) {
                    outf[(size_t)row * Nn + col] = vacc;
                } else if (MODE == 1) {
                    int b = row >> 10, ntok = row & 1023;
                    float val = vacc + bias[col] + resid[(size_t)row * 768 + col];
                    int idx = ntok * 768 + col;             // faithful untransposed reshape
                    int hh = idx >> 16, rem = idx & 65535;
                    int np = rem >> 6, dp = rem & 63;
                    val += qres[(((size_t)(b * 12 + hh)) * 1024 + np) * 64 + dp];
                    outf[(size_t)row * 768 + col] = val;
                } else if (MODE == 2) {
                    float g = vacc + bias[col];
                    g = 0.5f * g * (1.0f + erff(g * 0.70710678118654752f));
                    outb[(size_t)row * Nn + col] = (__bf16)g;
                } else {
                    size_t o = (size_t)row * 768 + col;
                    outf[o] += vacc + bias[col];
                }
            }
        }
    }
}

// --------------------- flash attention, 16 q-rows / wave -------------------
__device__ inline float hmax16(float v) {
#pragma unroll
    for (int m = 1; m < 16; m <<= 1) v = fmaxf(v, __shfl_xor(v, m, 32));
    return v;
}
__device__ inline float hsum16(float v) {
#pragma unroll
    for (int m = 1; m < 16; m <<= 1) v += __shfl_xor(v, m, 32);
    return v;
}

__global__ __launch_bounds__(32) void attn_fa(const float* __restrict__ q,
                                              const __bf16* __restrict__ k,
                                              const __bf16* __restrict__ v,
                                              __bf16* __restrict__ ctx) {
    __shared__ __bf16 pb[16 * 32];
    const int lane = threadIdx.x, lo = lane & 15, hi = lane >> 4;
    const int m0 = blockIdx.x * 16;
    const int h = blockIdx.y, b = blockIdx.z;
    const size_t bh = (size_t)(b * 12 + h);

    // q A-fragments (f32 -> bf16 on the fly), K = head_dim = 64 -> 2 frags
    v16bf qA[2];
#pragma unroll
    for (int f = 0; f < 2; f++) {
        const float* p0 = q + (bh * 1024 + m0 + lo) * 64 + 32 * f + 16 * hi;
        union { __bf16 e[16]; v16bf v; } t;
#pragma unroll
        for (int i = 0; i < 8; i++) { t.e[i] = (__bf16)p0[i]; t.e[8 + i] = (__bf16)p0[8 + i]; }
        qA[f] = t.v;
    }

    v8f cacc[4] = {};
    float mrow[8], lrow[8];
#pragma unroll
    for (int r = 0; r < 8; r++) { mrow[r] = -3.4e38f; lrow[r] = 0.f; }

    const __bf16* kbase = k + bh * 1024 * 64;
    const __bf16* vbase = v + bh * 64 * 1024;

    for (int j0 = 0; j0 < 1024; j0 += 32) {
        v8f s[2] = {};
#pragma unroll
        for (int nt = 0; nt < 2; nt++) {
#pragma unroll
            for (int f = 0; f < 2; f++) {
                const u32x4* kp = (const u32x4*)(kbase + (size_t)(j0 + 16 * nt + lo) * 64 + 32 * f + 16 * hi);
                BFrag kf; kf.u[0] = kp[0]; kf.u[1] = kp[1];
                s[nt] = WMMA_BF16(qA[f], kf.v, s[nt]);
            }
        }
        // online softmax update over 32 new columns
#pragma unroll
        for (int r = 0; r < 8; r++) {
            float a0 = s[0][r] * 0.125f, a1 = s[1][r] * 0.125f;
            float nm = fmaxf(mrow[r], hmax16(fmaxf(a0, a1)));
            float alpha = __expf(mrow[r] - nm);
            float p0 = __expf(a0 - nm), p1 = __expf(a1 - nm);
            lrow[r] = lrow[r] * alpha + hsum16(p0 + p1);
            mrow[r] = nm;
            s[0][r] = p0; s[1][r] = p1;
#pragma unroll
            for (int dt = 0; dt < 4; dt++) cacc[dt][r] *= alpha;
        }
        // stage P (C-layout) into LDS, read back as A-fragment (16x32 bf16)
#pragma unroll
        for (int nt = 0; nt < 2; nt++)
#pragma unroll
            for (int r = 0; r < 8; r++)
                pb[(r + 8 * hi) * 32 + 16 * nt + lo] = (__bf16)s[nt][r];
        __syncthreads();
        BFrag pA;
        {
            const u32x4* pp = (const u32x4*)(pb + (size_t)lo * 32 + 16 * hi);
            pA.u[0] = pp[0]; pA.u[1] = pp[1];
        }
        __syncthreads();
#pragma unroll
        for (int dt = 0; dt < 4; dt++) {
            const u32x4* vp = (const u32x4*)(vbase + (size_t)(16 * dt + lo) * 1024 + j0 + 16 * hi);
            BFrag vf; vf.u[0] = vp[0]; vf.u[1] = vp[1];
            cacc[dt] = WMMA_BF16(pA.v, vf.v, cacc[dt]);
        }
    }
    // normalize + write ctx as (B,N,H*D) bf16 (heads transposed correctly here)
#pragma unroll
    for (int r = 0; r < 8; r++) {
        float inv = 1.f / lrow[r];
        size_t o = ((size_t)b * 1024 + m0 + r + 8 * hi) * 768 + h * 64;
#pragma unroll
        for (int dt = 0; dt < 4; dt++)
            ctx[o + 16 * dt + lo] = (__bf16)(cacc[dt][r] * inv);
    }
}

// ---------------------------------------------------------------------------
extern "C" void kernel_launch(void* const* d_in, const int* in_sizes, int n_in,
                              void* d_out, int out_size, void* d_ws, size_t ws_size,
                              hipStream_t stream) {
    const float* before = (const float*)d_in[0];
    const float* after  = (const float*)d_in[1];
    const float* n1w = (const float*)d_in[2];
    const float* n1b = (const float*)d_in[3];
    const float* qkvw = (const float*)d_in[4];
    const float* hlnw = (const float*)d_in[5];
    const float* hlnb = (const float*)d_in[6];
    const float* projw = (const float*)d_in[7];
    const float* projb = (const float*)d_in[8];
    const float* n2w = (const float*)d_in[9];
    const float* n2b = (const float*)d_in[10];
    const float* fc1w = (const float*)d_in[11];
    const float* fc1b = (const float*)d_in[12];
    const float* fc2w = (const float*)d_in[13];
    const float* fc2b = (const float*)d_in[14];
    float* out = (float*)d_out;

    const size_t R = 8192, C = 768;
    char* ws = (char*)d_ws;
    size_t off = 0;
    auto alloc = [&](size_t bytes) -> char* {
        char* p = ws + off;
        off = (off + bytes + 255) & ~(size_t)255;
        return p;
    };
    __bf16* wqkv  = (__bf16*)alloc(2304 * 768 * 2);
    __bf16* wproj = (__bf16*)alloc(768 * 768 * 2);
    __bf16* wfc1  = (__bf16*)alloc(3072 * 768 * 2);
    __bf16* wfc2  = (__bf16*)alloc(768 * 3072 * 2);
    float*  xnf[2]; __bf16* xnb[2]; float* qf[2]; __bf16* kb[2]; __bf16* vb[2]; __bf16* ctxb[2];
    for (int s = 0; s < 2; s++) xnf[s] = (float*)alloc(R * C * 4);
    for (int s = 0; s < 2; s++) xnb[s] = (__bf16*)alloc(R * C * 2);
    for (int s = 0; s < 2; s++) qf[s]  = (float*)alloc(R * C * 4);
    for (int s = 0; s < 2; s++) kb[s]  = (__bf16*)alloc(R * C * 2);
    for (int s = 0; s < 2; s++) vb[s]  = (__bf16*)alloc(R * C * 2);
    for (int s = 0; s < 2; s++) ctxb[s] = (__bf16*)alloc(R * C * 2);
    float* qkvraw = (float*)alloc(R * 2304 * 4);
    __bf16* hbuf = (__bf16*)qkvraw;   // reuse: R*3072*2 <= R*2304*4, lifetimes disjoint

    // 1) weights -> bf16
    cvt_bf16<<<(2304 * 768 + 255) / 256, 256, 0, stream>>>(qkvw, wqkv, 2304 * 768);
    cvt_bf16<<<(768 * 768 + 255) / 256, 256, 0, stream>>>(projw, wproj, 768 * 768);
    cvt_bf16<<<(3072 * 768 + 255) / 256, 256, 0, stream>>>(fc1w, wfc1, 3072 * 768);
    cvt_bf16<<<(768 * 3072 + 255) / 256, 256, 0, stream>>>(fc2w, wfc2, 768 * 3072);

    // 2) LN1 on both streams
    ln768<<<(unsigned)R, 256, 0, stream>>>(before, n1w, n1b, xnf[0], xnb[0]);
    ln768<<<(unsigned)R, 256, 0, stream>>>(after,  n1w, n1b, xnf[1], xnb[1]);

    // 3) QKV GEMM + per-head LN (sequential, shared qkvraw scratch)
    for (int s = 0; s < 2; s++) {
        gemm32x64<0><<<dim3((unsigned)(R / 32), 2304 / 64), 32, 0, stream>>>(
            xnb[s], wqkv, (int)R, 2304, 768, qkvraw, nullptr, nullptr, nullptr, nullptr);
        head_ln<<<(unsigned)(R * 36 / 8), 256, 0, stream>>>(qkvraw, hlnw, hlnb, qf[s], kb[s], vb[s]);
    }

    // 4) cross attention: output stream s uses q of the OTHER stream, k/v of s
    for (int s = 0; s < 2; s++)
        attn_fa<<<dim3(1024 / 16, 12, 8), 32, 0, stream>>>(qf[1 - s], kb[s], vb[s], ctxb[s]);

    // 5) proj GEMM: out_s = gemm + proj_b + q_{1-s}.reshape + xn_s
    for (int s = 0; s < 2; s++)
        gemm32x64<1><<<dim3((unsigned)(R / 32), 768 / 64), 32, 0, stream>>>(
            ctxb[s], wproj, (int)R, 768, 768, out + (size_t)s * R * C, nullptr,
            projb, xnf[s], qf[1 - s]);

    // 6) MLP with LN2, exact GELU; += into d_out
    for (int s = 0; s < 2; s++) {
        ln768<<<(unsigned)R, 256, 0, stream>>>(out + (size_t)s * R * C, n2w, n2b, nullptr, xnb[0]);
        gemm32x64<2><<<dim3((unsigned)(R / 32), 3072 / 64), 32, 0, stream>>>(
            xnb[0], wfc1, (int)R, 3072, 768, nullptr, hbuf, fc1b, nullptr, nullptr);
        gemm32x64<3><<<dim3((unsigned)(R / 32), 768 / 64), 32, 0, stream>>>(
            hbuf, wfc2, (int)R, 768, 3072, out + (size_t)s * R * C, nullptr,
            fc2b, nullptr, nullptr);
    }
}
